// DecoderRNN_84954453115413
// MI455X (gfx1250) — matrix operand order repaired and checked by reading, not compiled, not run on hardware
//
#include <hip/hip_runtime.h>
#include <math.h>

// ---------------------------------------------------------------------------
// DecoderRNN for MI455X (gfx1250, wave32, WMMA).
//  - recurrence + MLP: exact fp32 WMMA (16x16x4), async-to-LDS staging
//  - vocab projection: split-bf16 (hi+lo) WMMA (16x16x32_bf16), 3 WMMAs per
//    k-step, fp32 accumulate -> near-fp32 accuracy at bf16 matrix throughput
// ---------------------------------------------------------------------------

typedef float  v2f   __attribute__((ext_vector_type(2)));
typedef float  v8f   __attribute__((ext_vector_type(8)));
typedef __bf16 v8bf  __attribute__((ext_vector_type(8)));
typedef __bf16 v16bf __attribute__((ext_vector_type(16)));

#define B_      32
#define L_      64
#define T_ENC   196
#define HID     512
#define VOCAB   32000
#define SCALE_  (1.0f / 22.6274169979695f)   // 1/sqrt(512)

__device__ __forceinline__ float sigf(float x) { return 1.0f / (1.0f + __expf(-x)); }

// ---- gfx1250 async global->LDS copy (ASYNCcnt) ----------------------------
__device__ __forceinline__ void async_copy16(const float* gsrc, float* ldst)
{
    unsigned lds = (unsigned)(unsigned long long)(void*)ldst;  // LDS byte offset
    asm volatile("global_load_async_to_lds_b128 %0, %1, off"
                 :: "v"(lds), "v"(gsrc)
                 : "memory");
}
__device__ __forceinline__ void async_wait0()
{
    asm volatile("s_wait_asynccnt 0" ::: "memory");
}

// ---------------------------------------------------------------------------
// Exact fp32 WMMA GEMM:  C[M,N] = act( A[M,K]*B[N,K]^T + bias0 + bias1 )
// Optional second (A1,B1,K1) pair accumulates into C (LSTM gates).
// Block 256 thr = 8 waves; block tile 32x64; one 16x16 tile per wave.
// M%32==0, N%64==0, K%64==0.
// ---------------------------------------------------------------------------
#define APAD 68   // floats per LDS row (64 + 4 pad)

__global__ __launch_bounds__(256) void gemm_wmma_f32(
    const float* __restrict__ A0, int lda0, int K0,
    const float* __restrict__ B0, int ldb0, const float* __restrict__ bias0,
    const float* __restrict__ A1, int lda1, int K1,
    const float* __restrict__ B1, int ldb1, const float* __restrict__ bias1,
    float* __restrict__ C, long ldc, int relu)
{
    __shared__ float sA[32 * APAD];
    __shared__ float sB[64 * APAD];

    const int tid  = threadIdx.x;
    const int lane = tid & 31;
    const int wave = tid >> 5;
    const int mt   = wave & 1;
    const int nt   = wave >> 1;
    const long rowBase = (long)blockIdx.x * 32;
    const long colBase = (long)blockIdx.y * 64;

    v8f acc = {};

    for (int part = 0; part < 2; ++part) {
        const float* A = part ? A1 : A0;
        const float* B = part ? B1 : B0;
        const int lda  = part ? lda1 : lda0;
        const int ldb  = part ? ldb1 : ldb0;
        const int K    = part ? K1   : K0;
        if (A == nullptr || K <= 0) continue;

        for (int k0 = 0; k0 < K; k0 += 64) {
            __syncthreads();   // everyone done reading previous chunk
            // ---- async-stage A chunk: 32x64 = 512 float4 (2/thread)
            #pragma unroll
            for (int i = 0; i < 2; ++i) {
                int f4 = tid + i * 256;
                int r  = f4 >> 4;
                int c4 = (f4 & 15) << 2;
                async_copy16(A + (rowBase + r) * (long)lda + k0 + c4,
                             &sA[r * APAD + c4]);
            }
            // ---- async-stage B chunk: 64x64 = 1024 float4 (4/thread)
            #pragma unroll
            for (int i = 0; i < 4; ++i) {
                int f4 = tid + i * 256;
                int r  = f4 >> 4;
                int c4 = (f4 & 15) << 2;
                async_copy16(B + (colBase + r) * (long)ldb + k0 + c4,
                             &sB[r * APAD + c4]);
            }
            async_wait0();
            __syncthreads();

            const int mr   = mt * 16 + (lane & 15);
            const int nr   = nt * 16 + (lane & 15);
            const int koff = (lane < 16) ? 0 : 2;
            #pragma unroll
            for (int kk = 0; kk < 64; kk += 4) {
                v2f a = *(const v2f*)&sA[mr * APAD + kk + koff];
                v2f b = *(const v2f*)&sB[nr * APAD + kk + koff];
                acc = __builtin_amdgcn_wmma_f32_16x16x4_f32(
                        false, a, false, b, (short)0, acc, false, false);
            }
        }
    }

    const long col = colBase + nt * 16 + (lane & 15);
    float bsum = (bias0 ? bias0[col] : 0.0f) + (bias1 ? bias1[col] : 0.0f);
    const int rAdd = (lane < 16) ? 0 : 8;
    #pragma unroll
    for (int v = 0; v < 8; ++v) {
        long row = rowBase + mt * 16 + v + rAdd;
        float x = acc[v] + bsum;
        if (relu) x = fmaxf(x, 0.0f);
        C[row * ldc + col] = x;
    }
}

// ---------------------------------------------------------------------------
// Split-bf16 projection GEMM:  C[M,N] = A[M,K]*B[N,K]^T + bias  (fp32 in/out)
// A,B split on the fly into bf16 hi/lo; D += Ah*Bh + Ah*Bl + Al*Bh.
// Block 256 thr = 8 waves; block tile 32x64; K chunk 64 (two 16x16x32 steps).
// ---------------------------------------------------------------------------
#define BPAD 72   // halves per LDS row (64 + 8 pad) -> conflict-free b128 reads

__device__ __forceinline__ void split_store4(float4 v,
                                             unsigned short* sh,
                                             unsigned short* sl)
{
    float a[4] = {v.x, v.y, v.z, v.w};
    unsigned short hh[4], ll[4];
    #pragma unroll
    for (int j = 0; j < 4; ++j) {
        unsigned u = __float_as_uint(a[j]);
        hh[j] = (unsigned short)(u >> 16);
        float rem = a[j] - __uint_as_float(u & 0xffff0000u);
        ll[j] = (unsigned short)(__float_as_uint(rem) >> 16);
    }
    *(uint2*)sh = make_uint2(hh[0] | ((unsigned)hh[1] << 16),
                             hh[2] | ((unsigned)hh[3] << 16));
    *(uint2*)sl = make_uint2(ll[0] | ((unsigned)ll[1] << 16),
                             ll[2] | ((unsigned)ll[3] << 16));
}

// A fragment (16x32 bf16): lane m, half h -> runs [ks+8h .. +7] and [ks+16+8h .. +7]
__device__ __forceinline__ v16bf load_a_frag(const unsigned short* s,
                                             int row, int ks, int half)
{
    v8bf r0 = *(const v8bf*)&s[row * BPAD + ks + half * 8];
    v8bf r1 = *(const v8bf*)&s[row * BPAD + ks + 16 + half * 8];
    return __builtin_shufflevector(r0, r1, 0,1,2,3,4,5,6,7,8,9,10,11,12,13,14,15);
}
// B fragment (32x16 bf16): lane n, half h -> runs [ks+8h .. +7] and [ks+16+8h .. +7]
__device__ __forceinline__ v16bf load_b_frag(const unsigned short* s,
                                             int row, int ks, int half)
{
    v8bf r0 = *(const v8bf*)&s[row * BPAD + ks + half * 8];
    v8bf r1 = *(const v8bf*)&s[row * BPAD + ks + 16 + half * 8];
    return __builtin_shufflevector(r0, r1, 0,1,2,3,4,5,6,7,8,9,10,11,12,13,14,15);
}

__global__ __launch_bounds__(256) void gemm_wmma_bf16split(
    const float* __restrict__ A, int lda,
    const float* __restrict__ B, int ldb,
    const float* __restrict__ bias,
    float* __restrict__ C, long ldc, int K)
{
    __shared__ unsigned short sAh[32 * BPAD];
    __shared__ unsigned short sAl[32 * BPAD];
    __shared__ unsigned short sBh[64 * BPAD];
    __shared__ unsigned short sBl[64 * BPAD];

    const int tid  = threadIdx.x;
    const int lane = tid & 31;
    const int wave = tid >> 5;
    const int mt   = wave & 1;
    const int nt   = wave >> 1;
    const long rowBase = (long)blockIdx.x * 32;
    const long colBase = (long)blockIdx.y * 64;
    const int half = lane >> 4;
    const int mr   = mt * 16 + (lane & 15);
    const int nr   = nt * 16 + (lane & 15);

    v8f acc = {};

    for (int k0 = 0; k0 < K; k0 += 64) {
        __syncthreads();
        // ---- stage+split A chunk: 32x64 fp32 (2 float4 / thread)
        #pragma unroll
        for (int i = 0; i < 2; ++i) {
            int f4 = tid + i * 256;
            int r  = f4 >> 4;
            int c4 = (f4 & 15) << 2;
            float4 v = *(const float4*)(A + (rowBase + r) * (long)lda + k0 + c4);
            split_store4(v, &sAh[r * BPAD + c4], &sAl[r * BPAD + c4]);
        }
        // ---- stage+split B chunk: 64x64 fp32 (4 float4 / thread)
        #pragma unroll
        for (int i = 0; i < 4; ++i) {
            int f4 = tid + i * 256;
            int r  = f4 >> 4;
            int c4 = (f4 & 15) << 2;
            float4 v = *(const float4*)(B + (colBase + r) * (long)ldb + k0 + c4);
            split_store4(v, &sBh[r * BPAD + c4], &sBl[r * BPAD + c4]);
        }
        __syncthreads();

        #pragma unroll
        for (int ks = 0; ks < 64; ks += 32) {
            v16bf ah = load_a_frag(sAh, mr, ks, half);
            v16bf al = load_a_frag(sAl, mr, ks, half);
            v16bf bh = load_b_frag(sBh, nr, ks, half);
            v16bf bl = load_b_frag(sBl, nr, ks, half);
            acc = __builtin_amdgcn_wmma_f32_16x16x32_bf16(
                    false, ah, false, bh, (short)0, acc, false, false);
            acc = __builtin_amdgcn_wmma_f32_16x16x32_bf16(
                    false, ah, false, bl, (short)0, acc, false, false);
            acc = __builtin_amdgcn_wmma_f32_16x16x32_bf16(
                    false, al, false, bh, (short)0, acc, false, false);
        }
    }

    const long col = colBase + nt * 16 + (lane & 15);
    const float bv = bias ? bias[col] : 0.0f;
    const int rAdd = (lane < 16) ? 0 : 8;
    #pragma unroll
    for (int v = 0; v < 8; ++v) {
        long row = rowBase + mt * 16 + v + rAdd;
        C[row * ldc + col] = acc[v] + bv;
    }
}

// ---------------------------------------------------------------------------
// Attention for one time step + embedding gather + concat into x_t.
// ---------------------------------------------------------------------------
__global__ __launch_bounds__(256) void attention_step(
    const float* __restrict__ h1,
    const float* __restrict__ enc,
    const float* __restrict__ embed,
    const int*   __restrict__ captions,
    int t,
    float* __restrict__ xt)
{
    __shared__ float q[HID];
    __shared__ float aw[256];
    __shared__ float red[256];

    const int b   = blockIdx.x;
    const int tid = threadIdx.x;

    for (int k = tid; k < HID; k += 256) q[k] = h1[b * HID + k];
    __syncthreads();

    float sc = -1e30f;
    if (tid < T_ENC) {
        const float* e = enc + ((long)b * T_ENC + tid) * HID;
        float d = 0.0f;
        for (int k = 0; k < HID; ++k) d += q[k] * e[k];
        sc = d * SCALE_;
    }
    red[tid] = sc;
    __syncthreads();
    for (int s = 128; s > 0; s >>= 1) {
        if (tid < s) red[tid] = fmaxf(red[tid], red[tid + s]);
        __syncthreads();
    }
    const float m = red[0];
    __syncthreads();
    float e = (tid < T_ENC) ? __expf(sc - m) : 0.0f;
    red[tid] = e;
    __syncthreads();
    for (int s = 128; s > 0; s >>= 1) {
        if (tid < s) red[tid] += red[tid + s];
        __syncthreads();
    }
    const float inv = 1.0f / red[0];
    __syncthreads();
    aw[tid] = e * inv;
    __syncthreads();

    for (int k = tid; k < HID; k += 256) {
        float acc = 0.0f;
        const float* eb = enc + (long)b * T_ENC * HID + k;
        for (int tt = 0; tt < T_ENC; ++tt) acc += aw[tt] * eb[(long)tt * HID];
        xt[b * 1024 + HID + k] = acc;
    }
    const int cap = captions[b * L_ + t];
    for (int k = tid; k < HID; k += 256)
        xt[b * 1024 + k] = embed[(long)cap * HID + k];
}

// ---------------------------------------------------------------------------
__global__ __launch_bounds__(256) void lstm_pointwise(
    const float* __restrict__ gates,
    float* __restrict__ h, float* __restrict__ c,
    float* __restrict__ hs_out)
{
    const int idx = blockIdx.x * 256 + threadIdx.x;
    const int b = idx >> 9;
    const int n = idx & 511;
    const float* g = gates + b * 2048;
    const float gi = sigf(g[n]);
    const float gf = sigf(g[n + 512]);
    const float gg = tanhf(g[n + 1024]);
    const float go = sigf(g[n + 1536]);
    const float cn = gf * c[idx] + gi * gg;
    const float hn = go * tanhf(cn);
    c[idx] = cn;
    h[idx] = hn;
    if (hs_out) hs_out[idx] = hn;
}

__global__ __launch_bounds__(256) void init_state(
    const float* __restrict__ f,
    float* h0, float* c0, float* h1, float* c1)
{
    const int idx = blockIdx.x * 256 + threadIdx.x;
    const float v = f[idx];
    h0[idx] = v; c0[idx] = v; h1[idx] = v; c1[idx] = v;
}

// ---------------------------------------------------------------------------
extern "C" void kernel_launch(void* const* d_in, const int* in_sizes, int n_in,
                              void* d_out, int out_size, void* d_ws, size_t ws_size,
                              hipStream_t stream)
{
    const float* features = (const float*)d_in[0];
    const int*   captions = (const int*)  d_in[1];
    const float* enc      = (const float*)d_in[3];
    const float* embed    = (const float*)d_in[4];
    const float* ft1_w    = (const float*)d_in[5];
    const float* ft1_b    = (const float*)d_in[6];
    const float* ft2_w    = (const float*)d_in[7];
    const float* ft2_b    = (const float*)d_in[8];
    const float* w_ih0    = (const float*)d_in[9];
    const float* w_hh0    = (const float*)d_in[10];
    const float* b_ih0    = (const float*)d_in[11];
    const float* b_hh0    = (const float*)d_in[12];
    const float* w_ih1    = (const float*)d_in[13];
    const float* w_hh1    = (const float*)d_in[14];
    const float* b_ih1    = (const float*)d_in[15];
    const float* b_hh1    = (const float*)d_in[16];
    const float* fc_w     = (const float*)d_in[17];
    const float* fc_b     = (const float*)d_in[18];
    float* out = (float*)d_out;

    float* ws = (float*)d_ws;
    float* f1    = ws;                 // 32*512
    float* f     = f1    + 16384;      // 32*512
    float* h0    = f     + 16384;
    float* c0    = h0    + 16384;
    float* h1    = c0    + 16384;
    float* c1    = h1    + 16384;
    float* xt    = c1    + 16384;      // 32*1024
    float* gates = xt    + 32768;      // 32*2048
    float* hs    = gates + 65536;      // 2048*512, row = t*32 + b

    gemm_wmma_f32<<<dim3(1, HID / 64), 256, 0, stream>>>(
        features, 2048, 2048, ft1_w, 2048, ft1_b,
        nullptr, 0, 0, nullptr, 0, nullptr,
        f1, HID, 1);
    gemm_wmma_f32<<<dim3(1, HID / 64), 256, 0, stream>>>(
        f1, HID, HID, ft2_w, HID, ft2_b,
        nullptr, 0, 0, nullptr, 0, nullptr,
        f, HID, 0);
    init_state<<<64, 256, 0, stream>>>(f, h0, c0, h1, c1);

    for (int t = 0; t < L_; ++t) {
        attention_step<<<32, 256, 0, stream>>>(h1, enc, embed, captions, t, xt);
        gemm_wmma_f32<<<dim3(1, 2048 / 64), 256, 0, stream>>>(
            xt, 1024, 1024, w_ih0, 1024, b_ih0,
            h0, HID, HID, w_hh0, HID, b_hh0,
            gates, 2048, 0);
        lstm_pointwise<<<64, 256, 0, stream>>>(gates, h0, c0, nullptr);
        gemm_wmma_f32<<<dim3(1, 2048 / 64), 256, 0, stream>>>(
            h0, HID, HID, w_ih1, HID, b_ih1,
            h1, HID, HID, w_hh1, HID, b_hh1,
            gates, 2048, 0);
        lstm_pointwise<<<64, 256, 0, stream>>>(gates, h1, c1, hs + (long)t * 16384);
    }

    // vocab projection in split-bf16 WMMA: [2048,512] x [32000,512]^T
    gemm_wmma_bf16split<<<dim3(2048 / 32, VOCAB / 64), 256, 0, stream>>>(
        hs, HID, fc_w, HID, fc_b, out, VOCAB, HID);
}
